// GLOBE_4054449127483
// MI455X (gfx1250) — compile-verified, matrix operand order checked by reading, not run on hardware
//
#include <hip/hip_runtime.h>
#include <math.h>

typedef _Float16 v16h __attribute__((ext_vector_type(16)));
typedef _Float16 v8h  __attribute__((ext_vector_type(8)));
typedef float    v8f  __attribute__((ext_vector_type(8)));

constexpr int   HD       = 64;                     // hidden width
constexpr float EPSV     = 1e-8f;
constexpr float FOURPI_F = 12.566370614359172f;

struct LayerArgs {
  const float* tgtA; int nTgtA; const float* tgtB; int nTgt;
  const float* srcC[2]; const float* srcN[2]; const float* srcA[2]; int nSrc[2];
  const float* refL;
  const float* W1; const float* b1;
  const float* W2; const float* b2;
  const float* W3; const float* b3;
  const float* Wo; const float* bo;
  const float* featS; const float* featV; int featOfs1;
  float* outS; float* outV; float* outFinal;
  const float* calibW; const float* calibB;
  int din; int nout; int nScal; int nVec; int hasFeat; int isFinal;
};

// Fast SiLU: x * sigmoid(x) using v_exp_f32 + v_rcp_f32 (no IEEE div sequence).
__device__ __forceinline__ float silu_fast(float x) {
  return x * __builtin_amdgcn_rcpf(1.f + __expf(-x));
}

// Load a 16x32 f16 WMMA A/B fragment from an LDS matrix stored row-major with
// row stride 64 halves.  Per CDNA5 ISA layout: lanes 0-15 hold K 0..7 (v0..3)
// and K 16..23 (v4..7); lanes 16-31 hold K 8..15 and K 24..31.
__device__ __forceinline__ v16h load_frag(const _Float16* base, int row, int kc, int hs) {
  const _Float16* p = base + row * HD + kc * 32 + hs * 8;
  v8h lo = *(const v8h*)(p);
  v8h hi = *(const v8h*)(p + 16);
  v16h a;
#pragma unroll
  for (int i = 0; i < 8; ++i) { a[i] = lo[i]; a[i + 8] = hi[i]; }
  return a;
}

__device__ __forceinline__ v8f wmma_k64(v16h a0, v16h a1, v16h b0, v16h b1) {
  v8f c = {};
  c = __builtin_amdgcn_wmma_f32_16x16x32_f16(false, a0, false, b0, (short)0, c, false, false);
  c = __builtin_amdgcn_wmma_f32_16x16x32_f16(false, a1, false, b1, (short)0, c, false, false);
  return c;
}

__launch_bounds__(256)
__global__ void fused_bem_layer(LayerArgs A) {
  // ---------------- LDS (≈51.5 KB static) ----------------
  __shared__ alignas(16) _Float16 wt1[HD * HD];     // W1^T  [N][K] f16, K zero-padded
  __shared__ alignas(16) _Float16 wt2[HD * HD];
  __shared__ alignas(16) _Float16 wt3[HD * HD];
  __shared__ alignas(16) _Float16 wto[32 * HD];     // Wo^T  [32][K]
  __shared__ float bs1[HD], bs2[HD], bs3[HD], bso[32];
  __shared__ alignas(16) _Float16 xbuf[128 * HD];   // activations, reused each stage
  __shared__ float wq[128];                         // per-pair weight a/(4*pi*d)
  __shared__ float rq[128 * 3];                     // per-pair rhat
  __shared__ float nq[128 * 3];                     // per-pair source normal
  __shared__ float accS[16 * 14];                   // per-target scalar accum
  __shared__ float accV[16 * 6 * 3];                // per-target vector accum

  const int tid     = threadIdx.x;
  const int lane    = tid & 31;
  const int wv      = tid >> 5;        // wave id 0..7  -> M-tile
  const int col     = lane & 15;       // N column within tile
  const int half    = lane >> 4;       // 0: rows 0..7, 1: rows 8..15
  const int rowBase = wv * 16;         // pair-row base of this wave's M-tile
  const int tgt0    = blockIdx.x * 16;

  for (int i = tid; i < 16 * 14; i += 256) accS[i] = 0.f;
  for (int i = tid; i < 16 * 6 * 3; i += 256) accV[i] = 0.f;
  __syncthreads();

  for (int bi = 0; bi < 2; ++bi) {
    for (int li = 0; li < 2; ++li) {
      const int pi = bi * 2 + li;

      // ---- stage this (bi,li)'s weights into LDS, transposed + f16 ----
      const float* W1p = A.W1 + (size_t)pi * A.din * HD;
      const float* W2p = A.W2 + (size_t)pi * HD * HD;
      const float* W3p = A.W3 + (size_t)pi * HD * HD;
      const float* Wop = A.Wo + (size_t)pi * HD * A.nout;
      for (int i = tid; i < HD * HD; i += 256) {
        int n = i >> 6, k = i & 63;
        wt1[n * HD + k] = (_Float16)((k < A.din) ? W1p[k * HD + n] : 0.f);
        wt2[n * HD + k] = (_Float16)W2p[k * HD + n];
        wt3[n * HD + k] = (_Float16)W3p[k * HD + n];
      }
      for (int i = tid; i < 32 * HD; i += 256) {
        int n = i >> 6, k = i & 63;
        wto[n * HD + k] = (_Float16)((n < A.nout) ? Wop[k * A.nout + n] : 0.f);
      }
      if (tid < HD) {
        bs1[tid] = A.b1[pi * HD + tid];
        bs2[tid] = A.b2[pi * HD + tid];
        bs3[tid] = A.b3[pi * HD + tid];
      }
      if (tid < 32) bso[tid] = (tid < A.nout) ? A.bo[pi * A.nout + tid] : 0.f;
      __syncthreads();

      const float Lref = A.refL[li];
      const float invL = __builtin_amdgcn_rcpf(Lref);
      const int   nS   = A.nSrc[bi];
      const float* sc  = A.srcC[bi];
      const float* sn  = A.srcN[bi];
      const float* sa  = A.srcA[bi];
      const int   fOfs = (bi == 0) ? 0 : A.featOfs1;
      const int nChunks = (nS + 7) / 8;

      for (int ch = 0; ch < nChunks; ++ch) {
        // ================= feature phase: 128 pairs =================
        if (tid < 128) {
          int p  = tid;
          int tl = p >> 3, sl = p & 7;
          int t  = tgt0 + tl; if (t >= A.nTgt) t = A.nTgt - 1;
          int s  = ch * 8 + sl;
          float valid = (s < nS) ? 1.f : 0.f;
          if (s >= nS) s = nS - 1;
          const float* tp = (t < A.nTgtA) ? (A.tgtA + 3 * t)
                                          : (A.tgtB + 3 * (t - A.nTgtA));
          float cx = sc[3 * s], cy = sc[3 * s + 1], cz = sc[3 * s + 2];
          float nx = sn[3 * s], ny = sn[3 * s + 1], nz = sn[3 * s + 2];
          float rx = (tp[0] - cx) * invL;
          float ry = (tp[1] - cy) * invL;
          float rz = (tp[2] - cz) * invL;
          float d2 = rx * rx + ry * ry + rz * rz + EPSV;
          float inv_d = __builtin_amdgcn_rsqf(d2);      // v_rsq_f32
          float rhx = rx * inv_d, rhy = ry * inv_d, rhz = rz * inv_d;
          float c = rhx * nx + rhy * ny + rhz * nz;
          _Float16* xr = &xbuf[p * HD];
          xr[0] = (_Float16)(0.5f * __logf(d2));        // log(d) = 0.5*log(d^2)
          xr[1] = (_Float16)1.f;
          xr[2] = (_Float16)c;
          xr[3] = (_Float16)(0.5f * (3.f * c * c - 1.f));
          xr[4] = (_Float16)(0.5f * (5.f * c * c * c - 3.f * c));
          int kf = 5;
          if (A.hasFeat) {
            int sfi = fOfs + s;
            const float* fsc = A.featS + sfi * 14;
#pragma unroll
            for (int k = 0; k < 14; ++k) xr[5 + k] = (_Float16)fsc[k];
            const float* fvc = A.featV + sfi * 18;
#pragma unroll
            for (int v = 0; v < 6; ++v) {
              float vx = fvc[3 * v], vy = fvc[3 * v + 1], vz = fvc[3 * v + 2];
              float m2 = vx * vx + vy * vy + vz * vz + EPSV;
              xr[19 + v] = (_Float16)(rhx * vx + rhy * vy + rhz * vz);
              xr[25 + v] = (_Float16)(nx * vx + ny * vy + nz * vz);
              xr[31 + v] = (_Float16)(m2 * __builtin_amdgcn_rsqf(m2)); // sqrt(m2)
            }
            kf = 37;
          }
          for (int k = kf; k < HD; ++k) xr[k] = (_Float16)0.f;
          wq[p] = valid * sa[s] * inv_d * (1.f / FOURPI_F);
          rq[3 * p] = rhx; rq[3 * p + 1] = rhy; rq[3 * p + 2] = rhz;
          nq[3 * p] = nx;  nq[3 * p + 1] = ny;  nq[3 * p + 2] = nz;
        }
        __syncthreads();

        // ================= hidden stages 1..3 (K=64, N=64) =================
        const _Float16* wts[3] = { wt1, wt2, wt3 };
        const float*    bss[3] = { bs1, bs2, bs3 };
#pragma unroll
        for (int st = 0; st < 3; ++st) {
          v16h a0 = load_frag(xbuf, rowBase + col, 0, half);
          v16h a1 = load_frag(xbuf, rowBase + col, 1, half);
          v8f acc[4];
#pragma unroll
          for (int nt = 0; nt < 4; ++nt) {
            v16h b0 = load_frag(wts[st], nt * 16 + col, 0, half);
            v16h b1 = load_frag(wts[st], nt * 16 + col, 1, half);
            acc[nt] = wmma_k64(a0, a1, b0, b1);
          }
          __syncthreads();   // all A-frag reads done before overwriting xbuf
#pragma unroll
          for (int nt = 0; nt < 4; ++nt) {
            float b = bss[st][nt * 16 + col];
#pragma unroll
            for (int v = 0; v < 8; ++v) {
              float x = silu_fast(acc[nt][v] + b);
              xbuf[(rowBase + v + 8 * half) * HD + nt * 16 + col] = (_Float16)x;
            }
          }
          __syncthreads();
        }

        // ============ output GEMM + weighted source reduction ============
        {
          v16h a0 = load_frag(xbuf, rowBase + col, 0, half);
          v16h a1 = load_frag(xbuf, rowBase + col, 1, half);
          const int NT = (A.nout + 15) >> 4;
          const int tl = 2 * wv + half;   // local target of this lane's rows
          for (int nt = 0; nt < NT; ++nt) {
            v16h b0 = load_frag(wto, nt * 16 + col, 0, half);
            v16h b1 = load_frag(wto, nt * 16 + col, 1, half);
            v8f cz = wmma_k64(a0, a1, b0, b1);
            const int chan = nt * 16 + col;
            const float b = bso[chan];
            if (chan < A.nScal) {
              float s0 = 0.f;
#pragma unroll
              for (int v = 0; v < 8; ++v) {
                int p = rowBase + v + 8 * half;
                s0 += wq[p] * (cz[v] + b);
              }
              accS[tl * 14 + chan] += s0;     // unique (tl,chan) per lane
            } else if (chan < A.nScal + A.nVec) {
              int vv = chan - A.nScal;        // av: project on rhat
              float s1 = 0.f, s2 = 0.f, s3 = 0.f;
#pragma unroll
              for (int v = 0; v < 8; ++v) {
                int p = rowBase + v + 8 * half;
                float whv = wq[p] * (cz[v] + b);
                s1 += whv * rq[3 * p];
                s2 += whv * rq[3 * p + 1];
                s3 += whv * rq[3 * p + 2];
              }
              atomicAdd(&accV[(tl * 6 + vv) * 3 + 0], s1);
              atomicAdd(&accV[(tl * 6 + vv) * 3 + 1], s2);
              atomicAdd(&accV[(tl * 6 + vv) * 3 + 2], s3);
            } else if (chan < A.nScal + 2 * A.nVec) {
              int vv = chan - A.nScal - A.nVec; // bv: project on source normal
              float s1 = 0.f, s2 = 0.f, s3 = 0.f;
#pragma unroll
              for (int v = 0; v < 8; ++v) {
                int p = rowBase + v + 8 * half;
                float whv = wq[p] * (cz[v] + b);
                s1 += whv * nq[3 * p];
                s2 += whv * nq[3 * p + 1];
                s3 += whv * nq[3 * p + 2];
              }
              atomicAdd(&accV[(tl * 6 + vv) * 3 + 0], s1);
              atomicAdd(&accV[(tl * 6 + vv) * 3 + 1], s2);
              atomicAdd(&accV[(tl * 6 + vv) * 3 + 2], s3);
            }
          }
        }
        __syncthreads();   // protect xbuf/wq/rq/nq before next chunk
      } // chunks
      __syncthreads();     // protect weight LDS before refill
    }
  }

  // ---------------- write-out ----------------
  __syncthreads();
  if (!A.isFinal) {
    for (int i = tid; i < 16 * 14; i += 256) {
      int tl = i / 14, k = i % 14;
      int t = tgt0 + tl;
      if (t < A.nTgt && k < A.nScal) A.outS[t * A.nScal + k] = accS[tl * 14 + k];
    }
    for (int i = tid; i < 16 * 18; i += 256) {
      int tl = i / 18, j = i % 18;
      int t = tgt0 + tl;
      if (t < A.nTgt && (j / 3) < A.nVec)
        A.outV[t * (A.nVec * 3) + j] = accV[tl * 18 + j];
    }
  } else {
    float cw0 = A.calibW[0], cw1 = A.calibW[1], cb = A.calibB[0];
    for (int i = tid; i < 16; i += 256) {
      int t = tgt0 + i;
      if (t < A.nTgt) {
        float* o = A.outFinal + (size_t)t * 4;
        o[0] = cw0 * accS[i * 14 + 0] + cb;
        o[1] = cw1 * accV[i * 18 + 0];
        o[2] = cw1 * accV[i * 18 + 1];
        o[3] = cw1 * accV[i * 18 + 2];
      }
    }
  }
}

extern "C" void kernel_launch(void* const* d_in, const int* in_sizes, int n_in,
                              void* d_out, int out_size, void* d_ws, size_t ws_size,
                              hipStream_t stream) {
  const float* pred = (const float*)d_in[0];
  const float* cns  = (const float*)d_in[1];
  const float* nns  = (const float*)d_in[2];
  const float* ans  = (const float*)d_in[3];
  const float* cfs  = (const float*)d_in[4];
  const float* nfs  = (const float*)d_in[5];
  const float* afs  = (const float*)d_in[6];
  const float* refL = (const float*)d_in[7];
  const int NP  = in_sizes[0] / 3;
  const int NS  = in_sizes[3];
  const int NF  = in_sizes[6];
  const int T01 = NS + NF;

  float* ws    = (float*)d_ws;
  float* scalA = ws;
  float* vecA  = scalA + (size_t)T01 * 14;
  float* scalB = vecA + (size_t)T01 * 18;
  float* vecB  = scalB + (size_t)T01 * 14;

  LayerArgs base{};
  base.srcC[0] = cns; base.srcN[0] = nns; base.srcA[0] = ans; base.nSrc[0] = NS;
  base.srcC[1] = cfs; base.srcN[1] = nfs; base.srcA[1] = afs; base.nSrc[1] = NF;
  base.refL = refL; base.featOfs1 = NS;
  base.calibW = (const float*)d_in[32];
  base.calibB = (const float*)d_in[33];

  auto fill = [&](LayerArgs& a, int wbase) {
    a.W1 = (const float*)d_in[wbase + 0]; a.b1 = (const float*)d_in[wbase + 1];
    a.W2 = (const float*)d_in[wbase + 2]; a.b2 = (const float*)d_in[wbase + 3];
    a.W3 = (const float*)d_in[wbase + 4]; a.b3 = (const float*)d_in[wbase + 5];
    a.Wo = (const float*)d_in[wbase + 6]; a.bo = (const float*)d_in[wbase + 7];
    a.din  = in_sizes[wbase + 0] / (4 * 64);  // (2,2,din,64)
    a.nout = in_sizes[wbase + 6] / (4 * 64);  // (2,2,64,nout)
  };

  // ---- layer 0: targets = concat(centers_ns, centers_fs), no input feats ----
  LayerArgs a0 = base;
  a0.tgtA = cns; a0.nTgtA = NS; a0.tgtB = cfs; a0.nTgt = T01;
  fill(a0, 8);
  a0.featS = nullptr; a0.featV = nullptr;
  a0.outS = scalA; a0.outV = vecA; a0.outFinal = nullptr;
  a0.nScal = 14; a0.nVec = 6; a0.hasFeat = 0; a0.isFinal = 0;
  fused_bem_layer<<<dim3((T01 + 15) / 16), dim3(256), 0, stream>>>(a0);

  // ---- layer 1: same targets, feats from layer 0 ----
  LayerArgs a1 = base;
  a1.tgtA = cns; a1.nTgtA = NS; a1.tgtB = cfs; a1.nTgt = T01;
  fill(a1, 16);
  a1.featS = scalA; a1.featV = vecA;
  a1.outS = scalB; a1.outV = vecB; a1.outFinal = nullptr;
  a1.nScal = 14; a1.nVec = 6; a1.hasFeat = 1; a1.isFinal = 0;
  fused_bem_layer<<<dim3((T01 + 15) / 16), dim3(256), 0, stream>>>(a1);

  // ---- final layer: targets = prediction points, feats from layer 1 ----
  LayerArgs af = base;
  af.tgtA = pred; af.nTgtA = NP; af.tgtB = pred; af.nTgt = NP;
  fill(af, 24);
  af.featS = scalB; af.featV = vecB;
  af.outS = nullptr; af.outV = nullptr; af.outFinal = (float*)d_out;
  af.nScal = 1; af.nVec = 1; af.hasFeat = 1; af.isFinal = 1;
  fused_bem_layer<<<dim3((NP + 15) / 16), dim3(256), 0, stream>>>(af);
}
